// GCN_26233660244215
// MI455X (gfx1250) — matrix operand order repaired and checked by reading, not compiled, not run on hardware
//
#include <hip/hip_runtime.h>
#include <math.h>

#define N_NODES   100000
#define N_EDGES   1000000
#define IN_DIM    128
#define HID       64
#define NUM_GRAPHS 2048

typedef __attribute__((ext_vector_type(16))) _Float16 v16h;
typedef __attribute__((ext_vector_type(8)))  float    v8f;

// ---------------------------------------------------------------- utilities
__global__ __launch_bounds__(256) void zero_f32(float* __restrict__ p, int n) {
  int i = blockIdx.x * blockDim.x + threadIdx.x;
  if (i < n) p[i] = 0.0f;
}

__global__ __launch_bounds__(256) void init_deg(float* __restrict__ deg, int n) {
  int i = blockIdx.x * blockDim.x + threadIdx.x;
  if (i < n) deg[i] = 1.0f;              // self-loop contributes 1 to every node
}

__global__ __launch_bounds__(256) void count_deg(const int* __restrict__ col,
                                                 float* __restrict__ deg, int e) {
  int i = blockIdx.x * blockDim.x + threadIdx.x;
  if (i < e) atomicAdd(&deg[col[i]], 1.0f);
}

__global__ __launch_bounds__(256) void deg_to_dinv(float* __restrict__ d, int n) {
  int i = blockIdx.x * blockDim.x + threadIdx.x;
  if (i < n) d[i] = rsqrtf(d[i]);        // deg >= 1 always (self-loops)
}

// ---------------------------------------------------------------- WMMA GEMM
// C[M,64] = A[M,KDIM] @ W[KDIM,64]  (fp32 in/out, f16 multiply, f32 accumulate)
// One wave per 16-row tile; wave covers all four 16-col tiles.
// W is staged transposed in LDS as f16 so each B fragment is a contiguous load.
template <int KDIM>
__global__ __launch_bounds__(128) void gemm_wmma(const float* __restrict__ A,
                                                 const float* __restrict__ W,
                                                 float* __restrict__ C, int M) {
  __shared__ _Float16 ldsWT[64 * KDIM];          // [n][k], k contiguous
  const int tid = threadIdx.x;
  for (int i = tid; i < KDIM * 64; i += 128) {   // W is [k][n] row-major
    int k = i >> 6, n = i & 63;
    ldsWT[n * KDIM + k] = (_Float16)W[i];
  }
  __syncthreads();

  const int lane  = tid & 31;
  const int tileM = blockIdx.x * 4 + (tid >> 5);
  if (tileM * 16 >= M) return;                   // wave-uniform: EXEC stays all-1s

  const int  m    = lane & 15;
  const bool hi   = lane >= 16;
  const float* arow = A + (size_t)(tileM * 16 + m) * KDIM;

  // A 16x32 layout (matches B): lanes 0-15 hold K = 0..15 ; lanes 16-31 hold K = 16..31
  const int kA0 = hi ? 16 : 0;
  // B 32x16 layout used here (matches A): lanes 0-15 hold K = 0..15 ; lanes 16-31 hold K = 16..31
  const int kB0 = hi ? 16 : 0;

  v8f acc[4] = {};                               // 4 n-tiles of 16 cols

#pragma unroll
  for (int ks = 0; ks < KDIM; ks += 32) {
    v16h a;
#pragma unroll
    for (int j = 0; j < 8; ++j) a[j]     = (_Float16)arow[ks + kA0 + j];
#pragma unroll
    for (int j = 0; j < 8; ++j) a[8 + j] = (_Float16)arow[ks + kA0 + 8 + j];

#pragma unroll
    for (int t = 0; t < 4; ++t) {
      const int n = t * 16 + (lane & 15);
      v16h b = *reinterpret_cast<const v16h*>(&ldsWT[n * KDIM + ks + kB0]); // 32B aligned
      acc[t] = __builtin_amdgcn_wmma_f32_16x16x32_f16(
          /*neg_a=*/false, a, /*neg_b=*/false, b,
          /*c_mod=*/(short)0, acc[t], /*reuse_a=*/false, /*reuse_b=*/false);
    }
  }

  // C/D layout: VGPR j -> M = j (+8 for lanes 16-31), N = lane&15
  const int mBase = hi ? 8 : 0;
#pragma unroll
  for (int t = 0; t < 4; ++t) {
    const int coln = t * 16 + (lane & 15);
#pragma unroll
    for (int j = 0; j < 8; ++j)
      C[(size_t)(tileM * 16 + mBase + j) * 64 + coln] = acc[t][j];
  }
}

// ---------------------------------------------------------------- aggregation
// s[col[e]][c] += dinv[row[e]] * hw[row[e]][c]   (one wave per edge, 2 ch/lane)
__global__ __launch_bounds__(256) void edge_scatter(const int* __restrict__ row,
                                                    const int* __restrict__ col,
                                                    const float* __restrict__ dinv,
                                                    const float* __restrict__ hw,
                                                    float* __restrict__ s, int nE) {
  int gid = blockIdx.x * blockDim.x + threadIdx.x;
  int e = gid >> 5;
  if (e >= nE) return;
  int c = (gid & 31) * 2;
  int r = row[e], t = col[e];                    // broadcast (wave-uniform) loads
  float w = dinv[r];
  float2 v = *reinterpret_cast<const float2*>(hw + (size_t)r * 64 + c);
  atomicAdd(s + (size_t)t * 64 + c,     w * v.x);
  atomicAdd(s + (size_t)t * 64 + c + 1, w * v.y);
}

// h[i][c] = relu( dinv[i]*s[i][c] + dinv[i]^2*hw[i][c] + bias[c] )   (in place in s)
__global__ __launch_bounds__(256) void finalize_relu(float* __restrict__ s,
                                                     const float* __restrict__ hw,
                                                     const float* __restrict__ dinv,
                                                     const float* __restrict__ bias,
                                                     int n) {
  int gid = blockIdx.x * blockDim.x + threadIdx.x;
  if (gid >= n * 64) return;
  int i = gid >> 6, c = gid & 63;
  float di = dinv[i];
  float v = di * s[gid] + di * di * hw[gid] + bias[c];
  s[gid] = v > 0.0f ? v : 0.0f;
}

// ---------------------------------------------------------------- pooling
__global__ __launch_bounds__(256) void attn_pool(const float* __restrict__ h,
                                                 const int* __restrict__ batch,
                                                 const float* __restrict__ aw,
                                                 const float* __restrict__ ab,
                                                 const float* __restrict__ mw,
                                                 const float* __restrict__ mb,
                                                 float* __restrict__ pooled, int n) {
  int gid  = blockIdx.x * blockDim.x + threadIdx.x;
  int node = gid >> 5;
  int lane = threadIdx.x & 31;
  if (node >= n) return;                         // wave-uniform
  int c = lane * 2;
  float2 v = *reinterpret_cast<const float2*>(h + (size_t)node * 64 + c);
  float pa = v.x * aw[c] + v.y * aw[c + 1];
  float pm = v.x * mw[c] + v.y * mw[c + 1];
#pragma unroll
  for (int off = 16; off > 0; off >>= 1) {
    pa += __shfl_xor(pa, off, 32);
    pm += __shfl_xor(pm, off, 32);
  }
  float score = (pa + ab[0]) * (1.0f / (1.0f + __expf(-(pm + mb[0]))));
  int g = batch[node];
  atomicAdd(pooled + (size_t)g * 64 + c,     v.x * score);
  atomicAdd(pooled + (size_t)g * 64 + c + 1, v.y * score);
}

__global__ __launch_bounds__(256) void out_proj(const float* __restrict__ pooled,
                                                const float* __restrict__ ow,
                                                const float* __restrict__ ob,
                                                float* __restrict__ out, int ng) {
  int gid  = blockIdx.x * blockDim.x + threadIdx.x;
  int g    = gid >> 5;
  int lane = threadIdx.x & 31;
  if (g >= ng) return;                           // wave-uniform
  int c = lane * 2;
  float2 v = *reinterpret_cast<const float2*>(pooled + (size_t)g * 64 + c);
  float p = v.x * ow[c] + v.y * ow[c + 1];
#pragma unroll
  for (int off = 16; off > 0; off >>= 1) p += __shfl_xor(p, off, 32);
  if (lane == 0) out[g] = p + ob[0];
}

// ---------------------------------------------------------------- launch
extern "C" void kernel_launch(void* const* d_in, const int* in_sizes, int n_in,
                              void* d_out, int out_size, void* d_ws, size_t ws_size,
                              hipStream_t stream) {
  const float* x   = (const float*)d_in[0];
  const int*   ei  = (const int*)d_in[1];
  const int*   row = ei;                 // edge_index[0] = sources
  const int*   col = ei + N_EDGES;       // edge_index[1] = targets
  const int*   bat = (const int*)d_in[2];
  const float* W1  = (const float*)d_in[3];
  const float* b1  = (const float*)d_in[4];
  const float* W2  = (const float*)d_in[5];
  const float* b2  = (const float*)d_in[6];
  const float* aw  = (const float*)d_in[7];
  const float* ab  = (const float*)d_in[8];
  const float* mw  = (const float*)d_in[9];
  const float* mb  = (const float*)d_in[10];
  const float* ow  = (const float*)d_in[11];
  const float* ob  = (const float*)d_in[12];
  float* out = (float*)d_out;

  // workspace layout (~52.3 MB): dinv | bufA | bufB | pooled
  char*  ws     = (char*)d_ws;
  float* dinv   = (float*)ws;                                  // N floats (512 KB slot)
  float* bufA   = (float*)(ws + 512 * 1024);                   // N*64 = 25.6 MB
  float* bufB   = bufA + (size_t)N_NODES * HID;                // N*64 = 25.6 MB
  float* pooled = bufB + (size_t)N_NODES * HID;                // 2048*64 = 0.5 MB

  const int NF = N_NODES * HID;                                // 6.4 M elements
  const int tiles  = (N_NODES + 15) / 16;                      // 6250 (exact)
  const int gemmBl = (tiles + 3) / 4;

  // 1) symmetric-norm degrees
  init_deg   <<<(N_NODES + 255) / 256, 256, 0, stream>>>(dinv, N_NODES);
  count_deg  <<<(N_EDGES + 255) / 256, 256, 0, stream>>>(col, dinv, N_EDGES);
  deg_to_dinv<<<(N_NODES + 255) / 256, 256, 0, stream>>>(dinv, N_NODES);

  // 2) layer 1: hw1 = x @ W1 (WMMA) ; aggregate ; relu
  gemm_wmma<IN_DIM><<<gemmBl, 128, 0, stream>>>(x, W1, bufA, N_NODES);
  zero_f32     <<<(NF + 255) / 256, 256, 0, stream>>>(bufB, NF);
  edge_scatter <<<(N_EDGES * 32 + 255) / 256, 256, 0, stream>>>(row, col, dinv, bufA, bufB, N_EDGES);
  finalize_relu<<<(NF + 255) / 256, 256, 0, stream>>>(bufB, bufA, dinv, b1, N_NODES);  // bufB = h1

  // 3) layer 2: hw2 = h1 @ W2 (WMMA) ; aggregate ; relu
  gemm_wmma<HID><<<gemmBl, 128, 0, stream>>>(bufB, W2, bufA, N_NODES);
  zero_f32     <<<(NF + 255) / 256, 256, 0, stream>>>(bufB, NF);
  edge_scatter <<<(N_EDGES * 32 + 255) / 256, 256, 0, stream>>>(row, col, dinv, bufA, bufB, N_EDGES);
  finalize_relu<<<(NF + 255) / 256, 256, 0, stream>>>(bufB, bufA, dinv, b2, N_NODES);  // bufB = h2

  // 4) attention pooling + output projection
  zero_f32 <<<(NUM_GRAPHS * HID + 255) / 256, 256, 0, stream>>>(pooled, NUM_GRAPHS * HID);
  attn_pool<<<(N_NODES * 32 + 255) / 256, 256, 0, stream>>>(bufB, bat, aw, ab, mw, mb, pooled, N_NODES);
  out_proj <<<(NUM_GRAPHS * 32 + 255) / 256, 256, 0, stream>>>(pooled, ow, ob, out, NUM_GRAPHS);
}
